// ContactNet_53523882443489
// MI455X (gfx1250) — matrix operand order repaired and checked by reading, not compiled, not run on hardware
//
#include <hip/hip_runtime.h>
#include <hip/hip_bf16.h>
#include <stdint.h>

// ContactNet forward for MI455X (gfx1250, wave32, WMMA).
// Dense 1x1-conv layers: bf16 WMMA GEMM (v_wmma_f32_16x16x32_bf16), 4-wave
// workgroups stage the 16 x cinp weight panel in LDS (ds_load_b128 A-frags),
// each wave register-blocks 16 cout x 64 points (4 accumulator sets) for
// ~16 FLOP/byte of HBM traffic and 4 independent WMMA chains of ILP.
// BN+bias folded into per-channel alpha/shift epilogue. Feature tensors are
// point-major (B, pts, ch), ch padded to a multiple of 32, so A/B fragments
// are contiguous 128-bit loads per the CDNA5 VGPR layouts (ISA 7.12.2).

#define BB   4
#define NPTS 20000
#define GPTS 1024

typedef __bf16 bf16_t;
typedef bf16_t v16bf __attribute__((ext_vector_type(16)));
typedef float  v8f   __attribute__((ext_vector_type(8)));

static __device__ __forceinline__ unsigned short f2b(float f) {
  unsigned u = __builtin_bit_cast(unsigned, f);
  unsigned r = u + 0x7fffu + ((u >> 16) & 1u);   // round-to-nearest-even
  return (unsigned short)(r >> 16);
}
static __device__ __forceinline__ float b2f(unsigned short h) {
  unsigned u = ((unsigned)h) << 16;
  return __builtin_bit_cast(float, u);
}

// ---------------- layer prep: fold BN/bias, convert weights to bf16 ---------
__global__ void k_prep(const float* __restrict__ w, const float* __restrict__ bia,
                       const float* __restrict__ gam, const float* __restrict__ bet,
                       const float* __restrict__ mea, const float* __restrict__ var,
                       unsigned short* __restrict__ Wb, float* __restrict__ alp,
                       float* __restrict__ shf, int cin, int cinp, int cout) {
  int co = blockIdx.x * blockDim.x + threadIdx.x;
  if (co >= cout) return;
  float a = 1.f, s = 0.f;
  if (gam) { a = gam[co] * rsqrtf(var[co] + 1e-5f); s = bet[co] - mea[co] * a; }
  if (bia) s += bia[co] * a;
  alp[co] = a; shf[co] = s;
  for (int ci = 0; ci < cinp; ++ci)
    Wb[(size_t)co * cinp + ci] = (ci < cin) ? f2b(w[(size_t)co * cin + ci]) : (unsigned short)0;
}

// ---------------- farthest point sampling (one block per batch) -------------
__global__ void k_fps(const float* __restrict__ xyz, float* __restrict__ mind,
                      int* __restrict__ inds, float* __restrict__ nxyz, int N, int np) {
  int b = blockIdx.x, t = threadIdx.x;
  const float* X = xyz + (size_t)b * N * 3;
  float* md = mind + (size_t)b * N;
  __shared__ float sv[256];
  __shared__ int   si[256];
  for (int i = t; i < N; i += 256) md[i] = 1e10f;
  __syncthreads();
  int cur = 0;
  for (int it = 0; it < np; ++it) {
    if (t == 0) {
      inds[b * np + it] = cur;
      nxyz[((size_t)b * np + it) * 3 + 0] = X[cur * 3 + 0];
      nxyz[((size_t)b * np + it) * 3 + 1] = X[cur * 3 + 1];
      nxyz[((size_t)b * np + it) * 3 + 2] = X[cur * 3 + 2];
    }
    float px = X[cur * 3], py = X[cur * 3 + 1], pz = X[cur * 3 + 2];
    float bv = -1.f; int bi = 0;
    for (int i = t; i < N; i += 256) {
      float dx = X[i * 3] - px, dy = X[i * 3 + 1] - py, dz = X[i * 3 + 2] - pz;
      float d = dx * dx + dy * dy + dz * dz;
      float m = md[i]; m = m < d ? m : d; md[i] = m;
      if (m > bv) { bv = m; bi = i; }
    }
    sv[t] = bv; si[t] = bi;
    __syncthreads();
    for (int s = 128; s > 0; s >>= 1) {
      if (t < s) {
        if (sv[t + s] > sv[t] || (sv[t + s] == sv[t] && si[t + s] < si[t])) {
          sv[t] = sv[t + s]; si[t] = si[t + s];
        }
      }
      __syncthreads();
    }
    cur = si[0];
    __syncthreads();
  }
}

// ---------------- ball query: first ns in-radius indices, pad w/ first ------
__global__ __launch_bounds__(32)
void k_ball(const float* __restrict__ xyz, const float* __restrict__ nxyz,
            int* __restrict__ idx, int N, int np, int ns, float r2) {
  int lane = threadIdx.x;
  int p = blockIdx.x, b = blockIdx.y;
  const float* X = xyz + (size_t)b * N * 3;
  float cx = nxyz[((size_t)b * np + p) * 3 + 0];
  float cy = nxyz[((size_t)b * np + p) * 3 + 1];
  float cz = nxyz[((size_t)b * np + p) * 3 + 2];
  int* out = idx + ((size_t)b * np + p) * ns;
  int cnt = 0, first = -1;
  for (int j0 = 0; j0 < N && cnt < ns; j0 += 32) {
    int j = j0 + lane;
    bool ok = false;
    if (j < N) {
      float dx = X[j * 3] - cx, dy = X[j * 3 + 1] - cy, dz = X[j * 3 + 2] - cz;
      ok = (dx * dx + dy * dy + dz * dz) < r2;
    }
    unsigned mm = (unsigned)__ballot(ok);
    if (ok) {
      int pos = cnt + __popc(mm & ((1u << lane) - 1u));
      if (pos < ns) out[pos] = j;
    }
    if (first < 0 && mm) first = j0 + (__ffs(mm) - 1);
    cnt += __popc(mm);
  }
  if (first < 0) first = 0;
  for (int s = cnt + lane; s < ns; s += 32) out[s] = first;
}

// ---------------- grouping: (xyz-center)/r ++ gathered feats, zero pad ------
__global__ void k_group(const float* __restrict__ xyz, const float* __restrict__ nxyz,
                        const unsigned short* __restrict__ feats, const int* __restrict__ idx,
                        unsigned short* __restrict__ Xo, int N, int np, int ns,
                        int C, int cinp, float invr) {
  int row = blockIdx.x * 64 + threadIdx.x;
  if (row >= np * ns) return;
  int b = blockIdx.y;
  int p = row / ns;
  int i = idx[((size_t)b * np + p) * ns + (row % ns)];
  unsigned short* o = Xo + ((size_t)b * np * ns + row) * cinp;
  o[0] = f2b((xyz[((size_t)b * N + i) * 3 + 0] - nxyz[((size_t)b * np + p) * 3 + 0]) * invr);
  o[1] = f2b((xyz[((size_t)b * N + i) * 3 + 1] - nxyz[((size_t)b * np + p) * 3 + 1]) * invr);
  o[2] = f2b((xyz[((size_t)b * N + i) * 3 + 2] - nxyz[((size_t)b * np + p) * 3 + 2]) * invr);
  int c = 3;
  if (feats) {
    const unsigned short* f = feats + ((size_t)b * N + i) * C;
    for (; c < 3 + C; ++c) o[c] = f[c - 3];
  }
  for (; c < cinp; ++c) o[c] = 0;
}

// ---------------- fused GEMM + BN + (ReLU): Y = act(alpha*(W X) + shift) ----
// 4 waves / block. Block stages W panel (16 x cinp bf16) into LDS; each wave
// computes a 16(cout) x 64(points) tile with 4 independent WMMA accumulators.
union FragU { v16bf v; unsigned int u[8]; };

__global__ __launch_bounds__(128)
void k_gemm(const unsigned short* __restrict__ Wb, const float* __restrict__ alp,
            const float* __restrict__ shf, const unsigned short* __restrict__ X,
            unsigned short* __restrict__ Y, int rows, int cinp, int cout, int relu) {
  __shared__ unsigned short sW[16 * 512];   // 16KB, cinp <= 512
  int tid  = threadIdx.x;
  int lane = tid & 31;
  int wave = tid >> 5;
  int half = (lane >> 4) & 1;
  int l16  = lane & 15;
  int mtile = blockIdx.y;
  int b     = blockIdx.z;

  // cooperative stage of contiguous 16 x cinp weight panel -> LDS
  const uint4* wp4 = reinterpret_cast<const uint4*>(Wb + (size_t)mtile * 16 * cinp);
  uint4* sw4 = reinterpret_cast<uint4*>(sW);
  for (int i = tid; i < 2 * cinp; i += 128) sw4[i] = wp4[i];
  __syncthreads();

  int wbase = blockIdx.x * 256 + wave * 64;   // this wave's first point
  if (wbase >= rows) return;                  // wave-uniform, after barrier

  const unsigned short* Xb = X + (size_t)b * rows * cinp;
  const unsigned short* xp[4];
  int valid[4];
#pragma unroll
  for (int j = 0; j < 4; ++j) {
    int p0 = wbase + j * 16;
    valid[j] = (p0 < rows);
    xp[j] = Xb + (size_t)((valid[j] ? p0 : 0) + l16) * cinp;  // clamp: safe loads
  }
  const unsigned short* aw = sW + l16 * cinp;

  v8f acc[4];
#pragma unroll
  for (int j = 0; j < 4; ++j) acc[j] = (v8f){0.f, 0.f, 0.f, 0.f, 0.f, 0.f, 0.f, 0.f};

  for (int k0 = 0; k0 < cinp; k0 += 32) {
    FragU fa;
    // A (16x32 bf16): VGPR r -> K=2r+16*half (16 contiguous bf16 per half)
    *reinterpret_cast<uint4*>(&fa.u[0]) = *reinterpret_cast<const uint4*>(aw + k0 + 16 * half);
    *reinterpret_cast<uint4*>(&fa.u[4]) = *reinterpret_cast<const uint4*>(aw + k0 + 16 * half + 8);
#pragma unroll
    for (int j = 0; j < 4; ++j) {
      FragU fb;
      // B (32x16 bf16): VGPR r -> K=2r+16*half (16 contiguous bf16 per half)
      *reinterpret_cast<uint4*>(&fb.u[0]) =
          *reinterpret_cast<const uint4*>(xp[j] + k0 + 16 * half);
      *reinterpret_cast<uint4*>(&fb.u[4]) =
          *reinterpret_cast<const uint4*>(xp[j] + k0 + 16 * half + 8);
      acc[j] = __builtin_amdgcn_wmma_f32_16x16x32_bf16(false, fa.v, false, fb.v,
                                                       (short)0, acc[j], false, false);
    }
    if (k0 + 32 < cinp) {
#pragma unroll
      for (int j = 0; j < 4; ++j) __builtin_prefetch(xp[j] + k0 + 32, 0, 1);
    }
  }

  // C/D layout: lane holds column n=l16, rows m = r + 8*half
  int mbase = mtile * 16 + 8 * half;
  float av[8], sv2[8];
#pragma unroll
  for (int r = 0; r < 8; ++r) { av[r] = alp[mbase + r]; sv2[r] = shf[mbase + r]; }
#pragma unroll
  for (int j = 0; j < 4; ++j) {
    if (!valid[j]) continue;   // wave-uniform
    unsigned int ow[4];
#pragma unroll
    for (int r = 0; r < 8; ++r) {
      float v = acc[j][r] * av[r] + sv2[r];
      if (relu) v = v > 0.f ? v : 0.f;
      unsigned short h = f2b(v);
      if (r & 1) ow[r >> 1] |= ((unsigned)h) << 16; else ow[r >> 1] = h;
    }
    unsigned short* yp = Y + ((size_t)b * rows + wbase + j * 16 + l16) * cout + mbase;
    *reinterpret_cast<uint4*>(yp) = make_uint4(ow[0], ow[1], ow[2], ow[3]);
  }
}

// ---------------- maxpool over nsample ----------------
__global__ void k_maxpool(const unsigned short* __restrict__ Y, unsigned short* __restrict__ F,
                          int np, int ns, int C) {
  int t = blockIdx.x * 256 + threadIdx.x;
  if (t >= np * C) return;
  int b = blockIdx.y, p = t / C, c = t % C;
  const unsigned short* y = Y + ((size_t)b * np * ns + (size_t)p * ns) * C + c;
  float m = -1e30f;
  for (int s = 0; s < ns; ++s) m = fmaxf(m, b2f(y[(size_t)s * C]));
  F[((size_t)b * np + p) * C + c] = f2b(m);
}

// ---------------- three nearest neighbors + interpolation weights ----------
__global__ void k_threenn(const float* __restrict__ u, const float* __restrict__ kn,
                          int* __restrict__ idx, float* __restrict__ w, int n1, int n2) {
  int i = blockIdx.x * 64 + threadIdx.x;
  if (i >= n1) return;
  int b = blockIdx.y;
  const float* U = u + ((size_t)b * n1 + i) * 3;
  float ux = U[0], uy = U[1], uz = U[2];
  const float* K = kn + (size_t)b * n2 * 3;
  float d0 = 1e30f, d1 = 1e30f, d2 = 1e30f;
  int   i0 = 0, i1 = 0, i2 = 0;
  for (int j = 0; j < n2; ++j) {
    float dx = K[j * 3] - ux, dy = K[j * 3 + 1] - uy, dz = K[j * 3 + 2] - uz;
    float d = dx * dx + dy * dy + dz * dz;
    if (d < d0)      { d2 = d1; i2 = i1; d1 = d0; i1 = i0; d0 = d; i0 = j; }
    else if (d < d1) { d2 = d1; i2 = i1; d1 = d;  i1 = j; }
    else if (d < d2) { d2 = d;  i2 = j; }
  }
  float w0 = 1.f / (sqrtf(fmaxf(d0, 0.f)) + 1e-8f);
  float w1 = 1.f / (sqrtf(fmaxf(d1, 0.f)) + 1e-8f);
  float w2 = 1.f / (sqrtf(fmaxf(d2, 0.f)) + 1e-8f);
  float sw = w0 + w1 + w2;
  size_t o = ((size_t)b * n1 + i) * 3;
  idx[o] = i0; idx[o + 1] = i1; idx[o + 2] = i2;
  w[o] = w0 / sw; w[o + 1] = w1 / sw; w[o + 2] = w2 / sw;
}

__global__ void k_interp(const unsigned short* __restrict__ f2, const int* __restrict__ idx,
                         const float* __restrict__ w, unsigned short* __restrict__ Xo,
                         int n1, int n2, int C2, int cinp, int coff) {
  int t = blockIdx.x * 256 + threadIdx.x;
  if (t >= n1 * C2) return;
  int b = blockIdx.y, i = t / C2, c = t % C2;
  const int*   id = idx + ((size_t)b * n1 + i) * 3;
  const float* ww = w + ((size_t)b * n1 + i) * 3;
  float s = 0.f;
  for (int k = 0; k < 3; ++k) s += ww[k] * b2f(f2[((size_t)b * n2 + id[k]) * C2 + c]);
  Xo[((size_t)b * n1 + i) * cinp + coff + c] = f2b(s);
}

__global__ void k_copycols(const unsigned short* __restrict__ f1, unsigned short* __restrict__ Xo,
                           int n1, int C1, int cinp, int coff) {
  int t = blockIdx.x * 256 + threadIdx.x;
  if (t >= n1 * C1) return;
  int b = blockIdx.y, i = t / C1, c = t % C1;
  Xo[((size_t)b * n1 + i) * cinp + coff + c] = f1[((size_t)b * n1 + i) * C1 + c];
}

__global__ void k_bcast(const unsigned short* __restrict__ gemb, unsigned short* __restrict__ Xo,
                        int n1, int C, int cinp, int coff) {
  int t = blockIdx.x * 256 + threadIdx.x;
  if (t >= n1 * C) return;
  int b = blockIdx.y, i = t / C, c = t % C;
  Xo[((size_t)b * n1 + i) * cinp + coff + c] = gemb[b * C + c];
}

__global__ void k_graspX(const float* __restrict__ gp, unsigned short* __restrict__ Xo,
                         int G, int cinp) {
  int g = blockIdx.x * 64 + threadIdx.x;
  if (g >= G) return;
  int b = blockIdx.y;
  unsigned short* o = Xo + ((size_t)b * G + g) * cinp;
  o[0] = f2b(gp[((size_t)b * G + g) * 3 + 0]);
  o[1] = f2b(gp[((size_t)b * G + g) * 3 + 1]);
  o[2] = f2b(gp[((size_t)b * G + g) * 3 + 2]);
  for (int c = 3; c < cinp; ++c) o[c] = 0;
}

__global__ void k_gmax(const unsigned short* __restrict__ Yg, unsigned short* __restrict__ gemb,
                       int G, int C) {
  int t = blockIdx.x * 64 + threadIdx.x;
  if (t >= BB * C) return;
  int b = t / C, c = t % C;
  float m = -1e30f;
  for (int g = 0; g < G; ++g) m = fmaxf(m, b2f(Yg[((size_t)b * G + g) * C + c]));
  gemb[b * C + c] = f2b(m);
}

__global__ void k_conv1(const unsigned short* __restrict__ Wb, const float* __restrict__ alp,
                        const float* __restrict__ shf, const unsigned short* __restrict__ X,
                        float* __restrict__ out, int rows, int cinp) {
  int n = blockIdx.x * 256 + threadIdx.x;
  if (n >= rows) return;
  int b = blockIdx.y;
  const unsigned short* x = X + ((size_t)b * rows + n) * cinp;
  float s = 0.f;
  for (int c = 0; c < cinp; ++c) s += b2f(Wb[c]) * b2f(x[c]);
  out[(size_t)b * rows + n] = s * alp[0] + shf[0];
}

__global__ void k_loss(const float* __restrict__ cm, const float* __restrict__ lb,
                       float* __restrict__ out, int N) {
  int b = blockIdx.x, t = threadIdx.x;
  __shared__ float sn[256], sd[256];
  float a = 0.f, c = 0.f;
  for (int i = t; i < N; i += 256) {
    float l = lb[(size_t)b * N + i];
    float d = cm[(size_t)b * N + i] - l;
    a += d * d * l; c += l;
  }
  sn[t] = a; sd[t] = c;
  __syncthreads();
  for (int s = 128; s > 0; s >>= 1) {
    if (t < s) { sn[t] += sn[t + s]; sd[t] += sd[t + s]; }
    __syncthreads();
  }
  if (t == 0) out[b] = sn[0] / (sd[0] + 1e-6f);
}

// =========================== host driver ====================================
struct Layer {
  const float *w, *b, *g, *be, *me, *va;
  int cin, cout, cinp, relu;
  unsigned short* Wb;
  float *alp, *shf;
};

extern "C" void kernel_launch(void* const* d_in, const int* in_sizes, int n_in,
                              void* d_out, int out_size, void* d_ws, size_t ws_size,
                              hipStream_t stream) {
  (void)in_sizes; (void)n_in; (void)out_size; (void)ws_size;
  const float* pc    = (const float*)d_in[0];  // (B,N,3)
  const float* gp    = (const float*)d_in[1];  // (B,G,3)
  const float* lab   = (const float*)d_in[2];  // (B,N)
  const float* labp  = (const float*)d_in[3];  // (B,N)

  // ---- params: JAX tree_flatten order (sorted dict keys at every level) ----
  int cur = 4;
  auto nx = [&]() { return (const float*)d_in[cur++]; };
  auto bn = [&](int cin, int cout, int bias) {
    Layer L{}; if (bias) L.b = nx();
    L.be = nx(); L.g = nx(); L.me = nx(); L.va = nx(); L.w = nx();
    L.cin = cin; L.cout = cout; L.cinp = (cin + 31) & ~31; L.relu = 1; return L;
  };
  auto cv = [&](int cin, int cout) {
    Layer L{}; L.b = nx(); L.w = nx();
    L.cin = cin; L.cout = cout; L.cinp = (cin + 31) & ~31; L.relu = 0; return L;
  };
  Layer fp1[2] = { bn(512, 256, 0), bn(256, 256, 0) };
  Layer fp2[2] = { bn(512, 256, 0), bn(256, 256, 0) };
  Layer fp3[2] = { bn(384, 256, 0), bn(256, 256, 0) };
  Layer gr[3]  = { bn(3, 64, 1), bn(64, 128, 1), cv(128, 128) };
  Layer hd[3]  = { bn(384, 256, 1), bn(256, 256, 1), cv(256, 1) };
  Layer hc[3]  = { bn(384, 256, 1), bn(256, 256, 1), cv(256, 1) };
  Layer sa1[3] = { bn(3, 64, 0), bn(64, 64, 0), bn(64, 128, 0) };
  Layer sa2[3] = { bn(131, 128, 0), bn(128, 128, 0), bn(128, 256, 0) };
  Layer sa3[3] = { bn(259, 128, 0), bn(128, 128, 0), bn(128, 256, 0) };
  Layer sa4[3] = { bn(259, 128, 0), bn(128, 128, 0), bn(128, 256, 0) };

  // ---- deterministic bump allocator over d_ws ----
  char* base = (char*)d_ws;
  size_t off = 0;
  auto alloc = [&](size_t bytes) -> void* {
    off = (off + 255) & ~(size_t)255;
    void* p = base + off; off += bytes; return p;
  };

  Layer* groups[] = { fp1, fp2, fp3, gr, hd, hc, sa1, sa2, sa3, sa4 };
  int    gcnt[]   = { 2, 2, 2, 3, 3, 3, 3, 3, 3, 3 };
  for (int gi = 0; gi < 10; ++gi)
    for (int li = 0; li < gcnt[gi]; ++li) {
      Layer& L = groups[gi][li];
      L.Wb  = (unsigned short*)alloc((size_t)L.cout * L.cinp * 2);
      L.alp = (float*)alloc((size_t)L.cout * 4);
      L.shf = (float*)alloc((size_t)L.cout * 4);
      k_prep<<<dim3((L.cout + 63) / 64), dim3(64), 0, stream>>>(
          L.w, L.b, L.g, L.be, L.me, L.va, L.Wb, L.alp, L.shf, L.cin, L.cinp, L.cout);
    }

  float* fps_mind = (float*)alloc((size_t)BB * NPTS * 4);
  int*   fps_inds = (int*)alloc((size_t)BB * 512 * 4);
  int*   ballidx  = (int*)alloc((size_t)BB * 512 * 64 * 4);
  int*   nn_idx   = (int*)alloc((size_t)BB * NPTS * 3 * 4);
  float* nn_w     = (float*)alloc((size_t)BB * NPTS * 3 * 4);
  float* s1x = (float*)alloc((size_t)BB * 512 * 3 * 4);
  float* s2x = (float*)alloc((size_t)BB * 256 * 3 * 4);
  float* s3x = (float*)alloc((size_t)BB * 128 * 3 * 4);
  float* s4x = (float*)alloc((size_t)BB * 64 * 3 * 4);
  unsigned short* s1f   = (unsigned short*)alloc((size_t)BB * 512 * 128 * 2);
  unsigned short* s2f   = (unsigned short*)alloc((size_t)BB * 256 * 256 * 2);
  unsigned short* s3f   = (unsigned short*)alloc((size_t)BB * 128 * 256 * 2);
  unsigned short* s4f   = (unsigned short*)alloc((size_t)BB * 64 * 256 * 2);
  unsigned short* fp1f  = (unsigned short*)alloc((size_t)BB * 128 * 256 * 2);
  unsigned short* fp2f  = (unsigned short*)alloc((size_t)BB * 256 * 256 * 2);
  unsigned short* seedf = (unsigned short*)alloc((size_t)BB * 512 * 256 * 2);
  unsigned short* gembb = (unsigned short*)alloc((size_t)BB * 128 * 2);
  float* cmap  = (float*)alloc((size_t)BB * NPTS * 4);
  float* cmapc = (float*)alloc((size_t)BB * NPTS * 4);
  unsigned short* G0 = (unsigned short*)alloc((size_t)BB * NPTS * 384 * 2);
  unsigned short* G1 = (unsigned short*)alloc((size_t)BB * NPTS * 256 * 2);
  unsigned short* G2 = (unsigned short*)alloc((size_t)BB * NPTS * 256 * 2);

  auto gemm = [&](const Layer& L, const unsigned short* X, unsigned short* Y, int rows) {
    dim3 g((rows + 255) / 256, L.cout / 16, BB);
    k_gemm<<<g, dim3(128), 0, stream>>>(L.Wb, L.alp, L.shf, X, Y, rows, L.cinp, L.cout, L.relu);
  };

  auto runSA = [&](const float* xyz, int Nsrc, const unsigned short* feats, int C,
                   int np, int ns, float radius, Layer* Ls,
                   float* nxyz, unsigned short* outF) {
    k_fps<<<dim3(BB), dim3(256), 0, stream>>>(xyz, fps_mind, fps_inds, nxyz, Nsrc, np);
    k_ball<<<dim3(np, BB), dim3(32), 0, stream>>>(xyz, nxyz, ballidx, Nsrc, np, ns, radius * radius);
    int rows = np * ns;
    k_group<<<dim3((rows + 63) / 64, BB), dim3(64), 0, stream>>>(
        xyz, nxyz, feats, ballidx, G0, Nsrc, np, ns, C, Ls[0].cinp, 1.f / radius);
    gemm(Ls[0], G0, G1, rows);
    gemm(Ls[1], G1, G2, rows);
    gemm(Ls[2], G2, G1, rows);
    int Cc = Ls[2].cout;
    k_maxpool<<<dim3((np * Cc + 255) / 256, BB), dim3(256), 0, stream>>>(G1, outF, np, ns, Cc);
  };

  auto runFP = [&](const float* xyz1, int n1, const float* xyz2, int n2,
                   const unsigned short* f1, int C1, const unsigned short* f2, int C2,
                   Layer* Ls, unsigned short* outF) {
    k_threenn<<<dim3((n1 + 63) / 64, BB), dim3(64), 0, stream>>>(xyz1, xyz2, nn_idx, nn_w, n1, n2);
    int cinp = Ls[0].cinp;
    k_interp<<<dim3((n1 * C2 + 255) / 256, BB), dim3(256), 0, stream>>>(
        f2, nn_idx, nn_w, G0, n1, n2, C2, cinp, 0);
    k_copycols<<<dim3((n1 * C1 + 255) / 256, BB), dim3(256), 0, stream>>>(
        f1, G0, n1, C1, cinp, C2);
    gemm(Ls[0], G0, G1, n1);
    gemm(Ls[1], G1, outF, n1);
  };

  // ---- set abstraction ----
  runSA(pc,  NPTS, nullptr, 0,   512, 64, 0.04f, sa1, s1x, s1f);
  runSA(s1x, 512,  s1f,     128, 256, 32, 0.10f, sa2, s2x, s2f);
  runSA(s2x, 256,  s2f,     256, 128, 16, 0.20f, sa3, s3x, s3f);
  runSA(s3x, 128,  s3f,     256, 64,  16, 0.30f, sa4, s4x, s4f);

  // ---- feature propagation ----
  runFP(s3x, 128, s4x, 64,  s3f, 256, s4f,  256, fp1, fp1f);
  runFP(s2x, 256, s3x, 128, s2f, 256, fp1f, 256, fp2, fp2f);
  runFP(s1x, 512, s2x, 256, s1f, 128, fp2f, 256, fp3, seedf);

  // ---- grasp embedding ----
  k_graspX<<<dim3((GPTS + 63) / 64, BB), dim3(64), 0, stream>>>(gp, G0, GPTS, gr[0].cinp);
  gemm(gr[0], G0, G1, GPTS);
  gemm(gr[1], G1, G2, GPTS);
  gemm(gr[2], G2, G1, GPTS);
  k_gmax<<<dim3((BB * 128 + 63) / 64), dim3(64), 0, stream>>>(G1, gembb, GPTS, 128);

  // ---- head input: [interp(seed), g_emb broadcast] over all N points ----
  k_threenn<<<dim3((NPTS + 63) / 64, BB), dim3(64), 0, stream>>>(pc, s1x, nn_idx, nn_w, NPTS, 512);
  k_interp<<<dim3((NPTS * 256 + 255) / 256, BB), dim3(256), 0, stream>>>(
      seedf, nn_idx, nn_w, G0, NPTS, 512, 256, 384, 0);
  k_bcast<<<dim3((NPTS * 128 + 255) / 256, BB), dim3(256), 0, stream>>>(
      gembb, G0, NPTS, 128, 384, 256);

  // ---- heads + losses ----
  gemm(hd[0], G0, G1, NPTS);
  gemm(hd[1], G1, G2, NPTS);
  k_conv1<<<dim3((NPTS + 255) / 256, BB), dim3(256), 0, stream>>>(
      hd[2].Wb, hd[2].alp, hd[2].shf, G2, cmap, NPTS, hd[2].cinp);
  gemm(hc[0], G0, G1, NPTS);
  gemm(hc[1], G1, G2, NPTS);
  k_conv1<<<dim3((NPTS + 255) / 256, BB), dim3(256), 0, stream>>>(
      hc[2].Wb, hc[2].alp, hc[2].shf, G2, cmapc, NPTS, hc[2].cinp);

  float* out = (float*)d_out;
  k_loss<<<dim3(BB), dim3(256), 0, stream>>>(cmap,  lab,  out,     NPTS);
  k_loss<<<dim3(BB), dim3(256), 0, stream>>>(cmapc, labp, out + 4, NPTS);
}
